// MLA_41515153883262
// MI455X (gfx1250) — matrix operand order repaired and checked by reading, not compiled, not run on hardware
//
#include <hip/hip_runtime.h>
#include <hip/hip_bf16.h>
#include <stdint.h>

// ---------------------------------------------------------------------------
// MLA forward for MI455X (gfx1250, wave32, WMMA).
//   1. cast x -> bf16 ; transpose+cast all weights -> bf16 [N][K]
//   2. kvraw = x @ w_kv            (WMMA bf16 GEMM, f32 out)
//   3. kv    = rmsnorm(kvraw)      -> bf16
//   4. kfull = kv @ w_k ; v = kv @ w_v ; qn = x @ w_qn ; qr = x @ w_qr
//   5. pack+RoPE: q,k -> [B][H][S][192] bf16 ; v -> [B][H][DV][S] bf16
//   6. causal flash attention, one wave per 16-query tile, WMMA bf16,
//      S^T = K @ Q^T trick so P is natively in A-fragment layout
//   7. out = O @ w_o               (WMMA bf16 GEMM, f32 out)
// GEMM uses double-buffered GLOBAL_LOAD_ASYNC_TO_LDS_B128 (ASYNCcnt) when the
// toolchain exposes the builtin, else the synchronous load/ds_store path.
// ---------------------------------------------------------------------------

typedef __attribute__((ext_vector_type(16))) __bf16 v16bf;
typedef __attribute__((ext_vector_type(8)))  __bf16 v8bf;
typedef __attribute__((ext_vector_type(8)))  float  v8f;
typedef __attribute__((ext_vector_type(4)))  int    v4i;

#define DEV static __device__ __forceinline__

#if defined(__has_builtin)
#if __has_builtin(__builtin_amdgcn_global_load_async_to_lds_b128)
#define ASYNC_LDS 1
#endif
#endif

#ifdef ASYNC_LDS
#if __has_builtin(__builtin_amdgcn_s_wait_asynccnt)
#define WAIT_ASYNC(n) __builtin_amdgcn_s_wait_asynccnt(n)
#else
#define WAIT_ASYNC(n) asm volatile("s_wait_asynccnt %0" ::"n"(n) : "memory")
#endif
// global bf16 src -> LDS bf16 dst, 16 bytes per lane, tracked by ASYNCcnt
DEV void async_b128(const __bf16* g, __bf16* l) {
  __builtin_amdgcn_global_load_async_to_lds_b128(
      (__attribute__((address_space(1))) v4i*)(g),
      (__attribute__((address_space(3))) v4i*)(l), 0, 0);
}
#endif

constexpr int Bb  = 2;
constexpr int Ss  = 2048;
constexpr int Dd  = 2048;
constexpr int Hh  = 16;
constexpr int DNn = 128;
constexpr int DRr = 64;
constexpr int DVv = 128;
constexpr int Rr  = 512;
constexpr int BS  = Bb * Ss;          // 4096
constexpr int DKk = DNn + DRr;        // 192 (per-head q/k dim)
constexpr int HK  = Hh * DKk;         // 3072
constexpr int HV  = Hh * DVv;         // 2048

DEV __bf16 f2bf(float f) {
  union { float f; uint32_t u; } v; v.f = f;
  uint32_t r = (v.u + 0x7FFFu + ((v.u >> 16) & 1u)) >> 16;   // RNE
  union { uint16_t s; __bf16 b; } o; o.s = (uint16_t)r;
  return o.b;
}

// A-fragment (16x32, 16-bit): lane half hv -> K runs [hv*8, hv*8+8) and
// [16+hv*8, 16+hv*8+8) of the lane's row (two 16B contiguous loads).
DEV v16bf load_a_frag(const __bf16* row, int k0, int hv) {
  v8bf lo = *(const v8bf*)(row + k0 + hv * 8);
  v8bf hi = *(const v8bf*)(row + k0 + 16 + hv * 8);
  v16bf r;
#pragma unroll
  for (int i = 0; i < 8; ++i) { r[i] = lo[i]; r[i + 8] = hi[i]; }
  return r;
}

DEV v8f wmma_bf(v16bf a, v16bf b, v8f c) {
  return __builtin_amdgcn_wmma_f32_16x16x32_bf16(
      false, a, false, b, (short)0, c, false, false);
}

// ---------------------------------------------------------------------------
// Elementwise / packing kernels
// ---------------------------------------------------------------------------

__global__ void cast_bf16_kernel(const float* __restrict__ in,
                                 __bf16* __restrict__ out, int n) {
  int i = blockIdx.x * blockDim.x + threadIdx.x;
  if (i < n) out[i] = f2bf(in[i]);
}

// in: fp32 [K][N] row-major  ->  out: bf16 [N][K] row-major
__global__ void transpose_cast_kernel(const float* __restrict__ in,
                                      __bf16* __restrict__ out, int K, int N) {
  int i = blockIdx.x * blockDim.x + threadIdx.x;
  if (i >= K * N) return;
  int k = i / N, n = i - k * N;
  out[(size_t)n * K + k] = f2bf(in[i]);
}

// rmsnorm over rows of [BS][R], bf16 output. One wave per row.
__global__ __launch_bounds__(256)
void rmsnorm_kernel(const float* __restrict__ in, const float* __restrict__ g,
                    __bf16* __restrict__ out) {
  int wid  = threadIdx.x >> 5;
  int lane = threadIdx.x & 31;
  int row  = blockIdx.x * 8 + wid;
  const float* r = in + (size_t)row * Rr;
  float ss = 0.f;
#pragma unroll
  for (int i = 0; i < Rr / 32; ++i) {
    float x = r[lane + i * 32];
    ss += x * x;
  }
#pragma unroll
  for (int m = 16; m >= 1; m >>= 1) ss += __shfl_xor(ss, m);
  float inv = rsqrtf(ss / (float)Rr + 1e-6f);
  __bf16* o = out + (size_t)row * Rr;
#pragma unroll
  for (int i = 0; i < Rr / 32; ++i) {
    int j = lane + i * 32;
    o[j] = f2bf(r[j] * inv * g[j]);
  }
}

// kfull fp32 [BS][H*192]  ->  kb bf16 [B][H][S][192], RoPE on d in [128,192)
__global__ void pack_k_kernel(const float* __restrict__ kf,
                              const float* __restrict__ freqs,
                              __bf16* __restrict__ kb) {
  int i = blockIdx.x * blockDim.x + threadIdx.x;
  if (i >= BS * Hh * DKk) return;
  int d  = i % DKk;
  int h  = (i / DKk) % Hh;
  int bs = i / (DKk * Hh);
  int s  = bs % Ss, b = bs / Ss;
  const float* src = kf + (size_t)bs * HK + h * DKk;
  float val;
  if (d < DNn) {
    val = src[d];
  } else {
    int j = d - DNn, p = j >> 1;
    float xr = src[DNn + 2 * p], xi = src[DNn + 2 * p + 1];
    float fr = freqs[s * (DRr / 2) + p];
    float c = __cosf(fr), sn = __sinf(fr);
    val = (j & 1) ? (xr * sn + xi * c) : (xr * c - xi * sn);
  }
  kb[(((size_t)b * Hh + h) * Ss + s) * DKk + d] = f2bf(val);
}

// qn fp32 [BS][H*128], qr fp32 [BS][H*64] -> qb bf16 [B][H][S][192] w/ RoPE
__global__ void pack_q_kernel(const float* __restrict__ qn,
                              const float* __restrict__ qr,
                              const float* __restrict__ freqs,
                              __bf16* __restrict__ qb) {
  int i = blockIdx.x * blockDim.x + threadIdx.x;
  if (i >= BS * Hh * DKk) return;
  int d  = i % DKk;
  int h  = (i / DKk) % Hh;
  int bs = i / (DKk * Hh);
  int s  = bs % Ss, b = bs / Ss;
  float val;
  if (d < DNn) {
    val = qn[(size_t)bs * HV + h * DNn + d];
  } else {
    int j = d - DNn, p = j >> 1;
    const float* src = qr + (size_t)bs * (Hh * DRr) + h * DRr;
    float xr = src[2 * p], xi = src[2 * p + 1];
    float fr = freqs[s * (DRr / 2) + p];
    float c = __cosf(fr), sn = __sinf(fr);
    val = (j & 1) ? (xr * sn + xi * c) : (xr * c - xi * sn);
  }
  qb[(((size_t)b * Hh + h) * Ss + s) * DKk + d] = f2bf(val);
}

// v fp32 [BS][H*128] -> vt bf16 [B][H][DV][S]   (per-head transposed)
__global__ void pack_v_kernel(const float* __restrict__ v,
                              __bf16* __restrict__ vt) {
  int i = blockIdx.x * blockDim.x + threadIdx.x;
  if (i >= BS * HV) return;
  int dv = i % DVv;
  int h  = (i / DVv) % Hh;
  int bs = i / HV;
  int s  = bs % Ss, b = bs / Ss;
  vt[(((size_t)b * Hh + h) * DVv + dv) * Ss + s] = f2bf(v[i]);
}

// ---------------------------------------------------------------------------
// bf16 WMMA GEMM:  C[M][N] (f32) = A[M][K] (bf16,row) @ Bt[N][K]^T (bf16,row)
// 128x128 block tile, 32-deep K steps, 8 waves each owning a 16x128 strip.
// Async path: double-buffered LDS tiles filled by GLOBAL_LOAD_ASYNC_TO_LDS,
// next tile's DMA overlaps current tile's WMMAs, fenced by s_wait_asynccnt.
// ---------------------------------------------------------------------------
__global__ __launch_bounds__(256)
void gemm_bf16_tn(const __bf16* __restrict__ A, const __bf16* __restrict__ Bt,
                  float* __restrict__ C, int M, int N, int K) {
  const int tid  = threadIdx.x;
  const int wid  = tid >> 5;
  const int lane = tid & 31;
  const int hv   = lane >> 4;
  const int lq   = lane & 15;
  const int m0   = blockIdx.x * 128;
  const int n0   = blockIdx.y * 128;

  v8f acc[8];
#pragma unroll
  for (int i = 0; i < 8; ++i) acc[i] = v8f{0.f, 0.f, 0.f, 0.f, 0.f, 0.f, 0.f, 0.f};

#ifdef ASYNC_LDS
  __shared__ __align__(32) __bf16 lsA[2][128][32];
  __shared__ __align__(32) __bf16 lsB[2][128][32];

  // Per thread: 4 async b128 copies per tile pair (A half-tiles + B half-tiles).
  const int e  = tid * 8;        // 8 bf16 = 16 bytes per copy
  const int r0 = e >> 5;         // rows 0..63 (second issue covers +64)
  const int c0 = e & 31;

  auto issue_tiles = [&](int k0, int buf) {
    async_b128(A + (size_t)(m0 + r0) * K + k0 + c0,       &lsA[buf][r0][c0]);
    async_b128(A + (size_t)(m0 + r0 + 64) * K + k0 + c0,  &lsA[buf][r0 + 64][c0]);
    async_b128(Bt + (size_t)(n0 + r0) * K + k0 + c0,      &lsB[buf][r0][c0]);
    async_b128(Bt + (size_t)(n0 + r0 + 64) * K + k0 + c0, &lsB[buf][r0 + 64][c0]);
  };

  issue_tiles(0, 0);
  int buf = 0;
  for (int k0 = 0; k0 < K; k0 += 32) {
    const bool more = (k0 + 32) < K;
    if (more) issue_tiles(k0 + 32, buf ^ 1);   // overlap next DMA with compute
    if (more) { WAIT_ASYNC(4); } else { WAIT_ASYNC(0); }  // current tile landed
    __syncthreads();

    v16bf af = load_a_frag(&lsA[buf][wid * 16 + lq][0], 0, hv);
    v16bf bfr[8];
#pragma unroll
    for (int nt = 0; nt < 8; ++nt)
      bfr[nt] = load_a_frag(&lsB[buf][nt * 16 + lq][0], 0, hv);
#pragma unroll
    for (int nt = 0; nt < 8; ++nt)             // back-to-back WMMA issue
      acc[nt] = wmma_bf(af, bfr[nt], acc[nt]);

    __syncthreads();                           // release buf for overwrite
    buf ^= 1;
  }
#else
  __shared__ __align__(32) __bf16 lsA[128][32];
  __shared__ __align__(32) __bf16 lsB[128][32];
  const int rA = tid >> 1;
  const int sg = (tid & 1) * 16;

  for (int k0 = 0; k0 < K; k0 += 32) {
    *(v16bf*)&lsA[rA][sg] = *(const v16bf*)(A + (size_t)(m0 + rA) * K + k0 + sg);
    *(v16bf*)&lsB[rA][sg] = *(const v16bf*)(Bt + (size_t)(n0 + rA) * K + k0 + sg);
    if (k0 + 32 < K) {
      __builtin_prefetch(A + (size_t)(m0 + rA) * K + k0 + 32 + sg, 0, 0);
      __builtin_prefetch(Bt + (size_t)(n0 + rA) * K + k0 + 32 + sg, 0, 0);
    }
    __syncthreads();

    v16bf af = load_a_frag(&lsA[wid * 16 + lq][0], 0, hv);
    v16bf bfr[8];
#pragma unroll
    for (int nt = 0; nt < 8; ++nt)
      bfr[nt] = load_a_frag(&lsB[nt * 16 + lq][0], 0, hv);
#pragma unroll
    for (int nt = 0; nt < 8; ++nt)
      acc[nt] = wmma_bf(af, bfr[nt], acc[nt]);
    __syncthreads();
  }
#endif

  const int crow0 = m0 + wid * 16;
#pragma unroll
  for (int nt = 0; nt < 8; ++nt) {
#pragma unroll
    for (int v = 0; v < 8; ++v) {
      C[(size_t)(crow0 + v + 8 * hv) * N + n0 + nt * 16 + lq] = acc[nt][v];
    }
  }
}

// ---------------------------------------------------------------------------
// Causal flash attention. One wave per (b, h, 16-query tile).
// Computes S^T = K @ Q^T so the softmax'd P lands directly in the
// A-fragment register layout for P @ V (no cross-lane transpose).
// ---------------------------------------------------------------------------
__global__ __launch_bounds__(32)
void attn_kernel(const __bf16* __restrict__ qbuf,
                 const __bf16* __restrict__ kbuf,
                 const __bf16* __restrict__ vt,
                 __bf16* __restrict__ ob) {
  const int lane = threadIdx.x & 31;
  const int hv   = lane >> 4;
  const int lq   = lane & 15;
  const int blk  = blockIdx.x;
  const int qt = blk & (Ss / 16 - 1);       // 128 tiles
  const int h  = (blk >> 7) & (Hh - 1);
  const int b  = blk >> 11;

  const size_t head = ((size_t)b * Hh + h) * Ss;
  const int qidx = qt * 16 + lq;            // this lane's query row

  // Q fragments (B-layout: N = query = lane, K = feature): 6 chunks of 32
  const __bf16* qrow = qbuf + (head + qt * 16 + lq) * DKk;
  v16bf qf[6];
#pragma unroll
  for (int c = 0; c < 6; ++c)
    qf[c] = load_a_frag(qrow, c * 32, hv);

  v8f o[8];
#pragma unroll
  for (int i = 0; i < 8; ++i) o[i] = v8f{0.f, 0.f, 0.f, 0.f, 0.f, 0.f, 0.f, 0.f};
  float m = -1e30f, l = 0.f;
  const float scale = 0.07216878364870322f;   // 1/sqrt(192)
  const int kend = qt * 16 + 15;

  for (int k0 = 0; k0 <= kend; k0 += 32) {
    v8f s0 = v8f{0.f, 0.f, 0.f, 0.f, 0.f, 0.f, 0.f, 0.f};
    v8f s1 = s0;
    const __bf16* krow0 = kbuf + (head + k0 + lq) * DKk;       // keys k0..k0+15
    const __bf16* krow1 = krow0 + 16 * DKk;                    // keys +16..+31
#pragma unroll
    for (int c = 0; c < 6; ++c) {
      v16bf a0 = load_a_frag(krow0, c * 32, hv);
      v16bf a1 = load_a_frag(krow1, c * 32, hv);
      s0 = wmma_bf(a0, qf[c], s0);                             // S^T tile 0
      s1 = wmma_bf(a1, qf[c], s1);                             // S^T tile 1
    }

    // Online softmax (per query = per lane; keys live on the VGPR axis).
    float p0[8], p1[8];
    float mp = -1e30f;
#pragma unroll
    for (int v = 0; v < 8; ++v) {
      int key0 = k0 + v + 8 * hv;
      int key1 = key0 + 16;
      float a = (key0 <= qidx) ? s0[v] * scale : -1e30f;
      float c = (key1 <= qidx) ? s1[v] * scale : -1e30f;
      p0[v] = a; p1[v] = c;
      mp = fmaxf(mp, fmaxf(a, c));
    }
    mp = fmaxf(mp, __shfl_xor(mp, 16));        // combine the two key-halves
    float mnew  = fmaxf(m, mp);
    float alpha = __expf(m - mnew);
    float lsum = 0.f;
#pragma unroll
    for (int v = 0; v < 8; ++v) {
      p0[v] = __expf(p0[v] - mnew);
      p1[v] = __expf(p1[v] - mnew);
      lsum += p0[v] + p1[v];
    }
    l = l * alpha + lsum;
    m = mnew;

    // P is already in A-fragment layout: idx 0..7 -> keys hv*8+0..7,
    // idx 8..15 -> keys 16+hv*8+0..7  (exactly what this lane holds).
    v16bf pa;
#pragma unroll
    for (int v = 0; v < 8; ++v) { pa[v] = f2bf(p0[v]); pa[v + 8] = f2bf(p1[v]); }

    // Per-output-row rescale factors (row M = v + 8*hv lives on lane M).
    float aM[8];
#pragma unroll
    for (int v = 0; v < 8; ++v) aM[v] = __shfl(alpha, 8 * hv + v);

    const size_t vhead = ((size_t)b * Hh + h) * DVv;
#pragma unroll
    for (int nt = 0; nt < 8; ++nt) {
#pragma unroll
      for (int v = 0; v < 8; ++v) o[nt][v] *= aM[v];
      // V B-fragment: N = dv = lane, K = key -> contiguous in vt row.
      v16bf bv = load_a_frag(vt + (vhead + nt * 16 + lq) * Ss, k0, hv);
      o[nt] = wmma_bf(pa, bv, o[nt]);
    }
  }

  float lt = l + __shfl_xor(l, 16);
  float linv[8];
#pragma unroll
  for (int v = 0; v < 8; ++v) linv[v] = 1.f / __shfl(lt, 8 * hv + v);

#pragma unroll
  for (int nt = 0; nt < 8; ++nt) {
#pragma unroll
    for (int v = 0; v < 8; ++v) {
      size_t row = (size_t)b * Ss + qt * 16 + v + 8 * hv;
      ob[row * HV + h * DVv + nt * 16 + lq] = f2bf(o[nt][v] * linv[v]);
    }
  }
}

// ---------------------------------------------------------------------------
// Launch
// ---------------------------------------------------------------------------
extern "C" void kernel_launch(void* const* d_in, const int* in_sizes, int n_in,
                              void* d_out, int out_size, void* d_ws, size_t ws_size,
                              hipStream_t stream) {
  (void)in_sizes; (void)n_in; (void)out_size; (void)ws_size;
  const float* x     = (const float*)d_in[0];
  const float* freqs = (const float*)d_in[1];
  const float* w_kv  = (const float*)d_in[2];
  const float* g_kv  = (const float*)d_in[3];
  const float* w_k   = (const float*)d_in[4];
  const float* w_v   = (const float*)d_in[5];
  const float* w_qn  = (const float*)d_in[6];
  const float* w_qr  = (const float*)d_in[7];
  const float* w_o   = (const float*)d_in[8];
  float* out = (float*)d_out;

  char* ws = (char*)d_ws;
  size_t off = 0;
  auto alloc = [&](size_t bytes) -> void* {
    void* p = ws + off;
    off += (bytes + 255) & ~(size_t)255;
    return p;
  };

  __bf16* xb    = (__bf16*)alloc((size_t)BS * Dd * 2);
  __bf16* wkvT  = (__bf16*)alloc((size_t)Rr * Dd * 2);
  __bf16* wkT   = (__bf16*)alloc((size_t)HK * Rr * 2);
  __bf16* wvT   = (__bf16*)alloc((size_t)HV * Rr * 2);
  __bf16* wqnT  = (__bf16*)alloc((size_t)HV * Dd * 2);
  __bf16* wqrT  = (__bf16*)alloc((size_t)(Hh * DRr) * Dd * 2);
  __bf16* woT   = (__bf16*)alloc((size_t)Dd * HV * 2);
  float*  kvraw = (float*)alloc((size_t)BS * Rr * 4);
  __bf16* kvb   = (__bf16*)alloc((size_t)BS * Rr * 2);
  float*  kf32  = (float*)alloc((size_t)BS * HK * 4);
  float*  v32   = (float*)alloc((size_t)BS * HV * 4);
  float*  qn32  = (float*)alloc((size_t)BS * HV * 4);
  float*  qr32  = (float*)alloc((size_t)BS * Hh * DRr * 4);
  __bf16* kb    = (__bf16*)alloc((size_t)Bb * Hh * Ss * DKk * 2);
  __bf16* qb    = (__bf16*)alloc((size_t)Bb * Hh * Ss * DKk * 2);
  __bf16* vt    = (__bf16*)alloc((size_t)Bb * Hh * DVv * Ss * 2);
  __bf16* ob    = (__bf16*)alloc((size_t)BS * HV * 2);

  const int T = 256;
  auto blocks = [&](long long n) { return (int)((n + T - 1) / T); };

  // 1. casts / transposes
  cast_bf16_kernel<<<blocks((long long)BS * Dd), T, 0, stream>>>(x, xb, BS * Dd);
  transpose_cast_kernel<<<blocks((long long)Dd * Rr), T, 0, stream>>>(w_kv, wkvT, Dd, Rr);
  transpose_cast_kernel<<<blocks((long long)Rr * HK), T, 0, stream>>>(w_k, wkT, Rr, HK);
  transpose_cast_kernel<<<blocks((long long)Rr * HV), T, 0, stream>>>(w_v, wvT, Rr, HV);
  transpose_cast_kernel<<<blocks((long long)Dd * HV), T, 0, stream>>>(w_qn, wqnT, Dd, HV);
  transpose_cast_kernel<<<blocks((long long)Dd * Hh * DRr), T, 0, stream>>>(w_qr, wqrT, Dd, Hh * DRr);
  transpose_cast_kernel<<<blocks((long long)HV * Dd), T, 0, stream>>>(w_o, woT, HV, Dd);

  // 2. kv projection + rmsnorm
  gemm_bf16_tn<<<dim3(BS / 128, Rr / 128), 256, 0, stream>>>(xb, wkvT, kvraw, BS, Rr, Dd);
  rmsnorm_kernel<<<BS / 8, 256, 0, stream>>>(kvraw, g_kv, kvb);

  // 3. k/v/q projections
  gemm_bf16_tn<<<dim3(BS / 128, HK / 128), 256, 0, stream>>>(kvb, wkT, kf32, BS, HK, Rr);
  gemm_bf16_tn<<<dim3(BS / 128, HV / 128), 256, 0, stream>>>(kvb, wvT, v32, BS, HV, Rr);
  gemm_bf16_tn<<<dim3(BS / 128, HV / 128), 256, 0, stream>>>(xb, wqnT, qn32, BS, HV, Dd);
  gemm_bf16_tn<<<dim3(BS / 128, (Hh * DRr) / 128), 256, 0, stream>>>(xb, wqrT, qr32, BS, Hh * DRr, Dd);

  // 4. RoPE + repack into attention layouts
  pack_k_kernel<<<blocks((long long)BS * Hh * DKk), T, 0, stream>>>(kf32, freqs, kb);
  pack_q_kernel<<<blocks((long long)BS * Hh * DKk), T, 0, stream>>>(qn32, qr32, freqs, qb);
  pack_v_kernel<<<blocks((long long)BS * HV), T, 0, stream>>>(v32, vt);

  // 5. flash attention: one wave per (b,h,16-query tile)
  attn_kernel<<<Bb * Hh * (Ss / 16), 32, 0, stream>>>(qb, kb, vt, ob);

  // 6. output projection
  gemm_bf16_tn<<<dim3(BS / 128, Dd / 128), 256, 0, stream>>>(ob, woT, out, BS, Dd, HV);
}
